// CPMMCodeModel_63866163691847
// MI455X (gfx1250) — matrix operand order repaired and checked by reading, not compiled, not run on hardware
//
#include <hip/hip_runtime.h>
#include <hip/hip_bf16.h>

typedef __bf16 bf16_t;
typedef __attribute__((ext_vector_type(16))) __bf16 v16bf;
typedef __attribute__((ext_vector_type(8)))  float  v8f;
typedef __attribute__((ext_vector_type(4)))  float  f32x4;

#define S_LEN  2048
#define D_DIM  512
#define V_DIM  32000
#define NSYM_  256
#define NVAL_  256
#define NSLOT_ 32
#define BATCH_ 4
#define CHUNK_ 128
#define NC_    16
#define DEXP_  1024
#define SCALE_D 0.04419417382415922f   // 1/sqrt(512)

// ---------------------------------------------------------------------------
// WMMA helpers (CDNA5: V_WMMA_F32_16X16X32_BF16, wave32)
// ---------------------------------------------------------------------------
union FragU { v16bf v; f32x4 q[2]; };

__device__ inline v8f wmma_bf16(v16bf a, v16bf b, v8f c) {
  return __builtin_amdgcn_wmma_f32_16x16x32_bf16(false, a, false, b, (short)0, c,
                                                 false, false);
}

// A fragment 16x32 (MxK), row-major source. Lane (m = lane&15, g = lane>>4)
// holds K = [8g..8g+7] and [16+8g..16+8g+7] -> two contiguous 16B loads.
__device__ inline v16bf load_a16x32(const bf16_t* tile, int ld, int lane) {
  int m = lane & 15, g = lane >> 4;
  const bf16_t* p = tile + (size_t)m * ld + 8 * g;
  FragU u;
  u.q[0] = *(const f32x4*)(p);
  u.q[1] = *(const f32x4*)(p + 16);
  return u.v;
}

// B fragment 32x16 (KxN), sourced from WT[n][k] (transposed weights).
// Lane (n = lane&15, g = lane>>4) holds K = [8g..8g+7] and [16+8g..16+8g+7] -> two 16B loads.
__device__ inline v16bf load_b32x16(const bf16_t* wt, int ld, int lane) {
  int n = lane & 15, g = lane >> 4;
  const bf16_t* p = wt + (size_t)n * ld + 8 * g;
  FragU u;
  u.q[0] = *(const f32x4*)(p);
  u.q[1] = *(const f32x4*)(p + 16);
  return u.v;
}

__device__ inline float gelu_f(float x) {
  float x3 = x * x * x;
  return 0.5f * x * (1.0f + tanhf(0.7978845608028654f * (x + 0.044715f * x3)));
}

// ---------------------------------------------------------------------------
// Weight transpose + bf16 convert:  src (K x N) f32 -> dst (N x K) bf16
// ---------------------------------------------------------------------------
__global__ void transpose_bf16_kernel(const float* __restrict__ src,
                                      bf16_t* __restrict__ dst, int K, int N) {
  size_t i = (size_t)blockIdx.x * 256 + threadIdx.x;
  size_t total = (size_t)K * N;
  if (i < total) {
    int nn = (int)(i / K);
    int kk = (int)(i % K);
    dst[i] = (bf16_t)src[(size_t)kk * N + nn];
  }
}

// ---------------------------------------------------------------------------
// Embedding: x[b,s,:] = token_embed[tok] + pos_embed[s]  -> bf16
// ---------------------------------------------------------------------------
__global__ __launch_bounds__(256) void embed_kernel(const int* __restrict__ tok,
                                                    const float* __restrict__ te,
                                                    const float* __restrict__ pe,
                                                    bf16_t* __restrict__ xb) {
  size_t r = blockIdx.x;             // 0 .. B*S-1
  int s = (int)(r & (S_LEN - 1));
  int id = tok[r];
  const float* trow = te + (size_t)id * D_DIM;
  const float* prow = pe + (size_t)s * D_DIM;
  for (int d = threadIdx.x; d < D_DIM; d += 256)
    xb[r * D_DIM + d] = (bf16_t)(trow[d] + prow[d]);
}

// ---------------------------------------------------------------------------
// Conv-as-GEMM (implicit im2col over 5 taps, SAME padding), WMMA bf16.
// Each wave: 16 rows x 64 cols.  Block: 8 waves = 128 rows x 64 cols.
// ---------------------------------------------------------------------------
__global__ __launch_bounds__(256) void conv_gemm_kernel(const bf16_t* __restrict__ X,
                                                        const bf16_t* __restrict__ WT,
                                                        const float* __restrict__ bias,
                                                        float* __restrict__ outF,
                                                        bf16_t* __restrict__ outB,
                                                        int Cin, int Cout, int do_gelu) {
  const int wave = threadIdx.x >> 5, lane = threadIdx.x & 31;
  const int m0 = blockIdx.x * 128 + wave * 16;
  const int n0 = blockIdx.y * 64;
  const int mrow = m0 + (lane & 15);
  const int bidx = mrow >> 11;           // S = 2048
  const int spos = mrow & (S_LEN - 1);
  const int g = lane >> 4;

  v8f acc0 = {0,0,0,0,0,0,0,0};
  v8f acc1 = acc0, acc2 = acc0, acc3 = acc0;

  for (int t = 0; t < 5; ++t) {
    int sp = spos + t - 2;
    bool valid = (sp >= 0) && (sp < S_LEN);
    const bf16_t* arow = X + ((size_t)(bidx << 11) + (size_t)(valid ? sp : 0)) * Cin + 8 * g;
    const bf16_t* wt_t = WT + (size_t)t * Cout * Cin;
    for (int k0 = 0; k0 < Cin; k0 += 32) {
      FragU au;
      if (valid) {
        au.q[0] = *(const f32x4*)(arow + k0);
        au.q[1] = *(const f32x4*)(arow + k0 + 16);
      } else {
        f32x4 z = {0.f, 0.f, 0.f, 0.f};
        au.q[0] = z; au.q[1] = z;
      }
      v16bf b0 = load_b32x16(wt_t + (size_t)(n0 +  0) * Cin + k0, Cin, lane);
      v16bf b1 = load_b32x16(wt_t + (size_t)(n0 + 16) * Cin + k0, Cin, lane);
      v16bf b2 = load_b32x16(wt_t + (size_t)(n0 + 32) * Cin + k0, Cin, lane);
      v16bf b3 = load_b32x16(wt_t + (size_t)(n0 + 48) * Cin + k0, Cin, lane);
      acc0 = wmma_bf16(au.v, b0, acc0);
      acc1 = wmma_bf16(au.v, b1, acc1);
      acc2 = wmma_bf16(au.v, b2, acc2);
      acc3 = wmma_bf16(au.v, b3, acc3);
    }
  }

  const int n = lane & 15;
  #pragma unroll
  for (int j = 0; j < 4; ++j) {
    v8f acc = (j == 0) ? acc0 : (j == 1) ? acc1 : (j == 2) ? acc2 : acc3;
    int col = n0 + 16 * j + n;
    float bcol = bias[col];
    #pragma unroll
    for (int i = 0; i < 8; ++i) {
      int row = m0 + 8 * g + i;
      float val = acc[i] + bcol;
      if (do_gelu) val = gelu_f(val);
      if (outF) outF[(size_t)row * Cout + col] = val;
      if (outB) outB[(size_t)row * Cout + col] = (bf16_t)val;
    }
  }
}

// ---------------------------------------------------------------------------
// Plain GEMM + bias (lm_head): out[M,N] = A[M,K] @ WT[N,K]^T + bias
// ---------------------------------------------------------------------------
__global__ __launch_bounds__(256) void gemm_lm_kernel(const bf16_t* __restrict__ A,
                                                      const bf16_t* __restrict__ WT,
                                                      const float* __restrict__ bias,
                                                      float* __restrict__ out,
                                                      int M, int N, int K) {
  const int wave = threadIdx.x >> 5, lane = threadIdx.x & 31;
  const int m0 = blockIdx.x * 128 + wave * 16;
  const int n0 = blockIdx.y * 64;
  const bf16_t* a_base = A + (size_t)m0 * K;

  v8f acc0 = {0,0,0,0,0,0,0,0};
  v8f acc1 = acc0, acc2 = acc0, acc3 = acc0;

  for (int k0 = 0; k0 < K; k0 += 32) {
    __builtin_prefetch(WT + (size_t)(n0 + (lane & 15)) * K + k0 + 128, 0, 0);
    v16bf a  = load_a16x32(a_base + k0, K, lane);
    v16bf b0 = load_b32x16(WT + (size_t)(n0 +  0) * K + k0, K, lane);
    v16bf b1 = load_b32x16(WT + (size_t)(n0 + 16) * K + k0, K, lane);
    v16bf b2 = load_b32x16(WT + (size_t)(n0 + 32) * K + k0, K, lane);
    v16bf b3 = load_b32x16(WT + (size_t)(n0 + 48) * K + k0, K, lane);
    acc0 = wmma_bf16(a, b0, acc0);
    acc1 = wmma_bf16(a, b1, acc1);
    acc2 = wmma_bf16(a, b2, acc2);
    acc3 = wmma_bf16(a, b3, acc3);
  }

  const int n = lane & 15, g = lane >> 4;
  #pragma unroll
  for (int j = 0; j < 4; ++j) {
    v8f acc = (j == 0) ? acc0 : (j == 1) ? acc1 : (j == 2) ? acc2 : acc3;
    int col = n0 + 16 * j + n;
    float bcol = bias[col];
    #pragma unroll
    for (int i = 0; i < 8; ++i) {
      int row = m0 + 8 * g + i;
      out[(size_t)row * N + col] = acc[i] + bcol;
    }
  }
}

// ---------------------------------------------------------------------------
// LayerNorm over D=512 per row -> f32 + bf16 outputs
// ---------------------------------------------------------------------------
__device__ inline float blk_sum(float v, float* s_red) {
  int t = threadIdx.x;
  s_red[t] = v; __syncthreads();
  for (int off = 128; off > 0; off >>= 1) {
    if (t < off) s_red[t] += s_red[t + off];
    __syncthreads();
  }
  float r = s_red[0]; __syncthreads();
  return r;
}
__device__ inline float blk_max(float v, float* s_red) {
  int t = threadIdx.x;
  s_red[t] = v; __syncthreads();
  for (int off = 128; off > 0; off >>= 1) {
    if (t < off) s_red[t] = fmaxf(s_red[t], s_red[t + off]);
    __syncthreads();
  }
  float r = s_red[0]; __syncthreads();
  return r;
}

__global__ __launch_bounds__(256) void ln_kernel(const float* __restrict__ x,
                                                 const float* __restrict__ g,
                                                 const float* __restrict__ bb,
                                                 float* __restrict__ outF,
                                                 bf16_t* __restrict__ outB) {
  __shared__ float s_red[256];
  size_t r = blockIdx.x;
  const float* row = x + r * D_DIM;
  int t = threadIdx.x;
  float v0 = row[t], v1 = row[t + 256];
  float m = blk_sum(v0 + v1, s_red) * (1.0f / D_DIM);
  float d0 = v0 - m, d1 = v1 - m;
  float var = blk_sum(d0 * d0 + d1 * d1, s_red) * (1.0f / D_DIM);
  float inv = rsqrtf(var + 1e-6f);
  float y0 = d0 * inv * g[t] + bb[t];
  float y1 = d1 * inv * g[t + 256] + bb[t + 256];
  outF[r * D_DIM + t] = y0;
  outF[r * D_DIM + t + 256] = y1;
  outB[r * D_DIM + t] = (bf16_t)y0;
  outB[r * D_DIM + t + 256] = (bf16_t)y1;
}

// ---------------------------------------------------------------------------
// Graph memory read (shared by chunk recurrence and final readout)
// ---------------------------------------------------------------------------
__device__ void graph_call_dev(const float* mm, const float* smem_,
                               int idx, bool mask,
                               const float* rp_w, const float* rp_b,
                               float* s_h, float* s_attn, float* s_red,
                               float* s_fb) {
  int t = threadIdx.x;
  for (int d = t; d < D_DIM; d += 256) s_h[d] = mask ? mm[(size_t)idx * D_DIM + d] : 0.f;
  __syncthreads();
  for (int gs = 0; gs < 2; ++gs) {
    {
      float a = 0.f;
      const float* r = smem_ + (size_t)t * D_DIM;   // v = t (NVAL == blockDim)
      for (int k = 0; k < D_DIM; ++k) a += s_h[k] * r[k];
      s_attn[t] = a * SCALE_D;
    }
    __syncthreads();
    float mx = blk_max(s_attn[t], s_red);
    float e = __expf(s_attn[t] - mx);
    float sum = blk_sum(e, s_red);
    s_attn[t] = e / sum;
    __syncthreads();
    for (int d = t; d < D_DIM; d += 256) {
      float a = 0.f;
      for (int v = 0; v < NVAL_; ++v) a += s_attn[v] * smem_[(size_t)v * D_DIM + d];
      s_h[d] += a;
    }
    __syncthreads();
  }
  for (int d = t; d < D_DIM; d += 256) {
    float a = rp_b[d];
    for (int k = 0; k < D_DIM; ++k) a += s_h[k] * rp_w[(size_t)k * D_DIM + d];
    s_fb[d] = a;
  }
  __syncthreads();
}

// ---------------------------------------------------------------------------
// Fused 16-chunk recurrence. One block per batch (the recurrence is strictly
// sequential in ci but fully independent per batch).
// ---------------------------------------------------------------------------
struct ChunkP {
  const float* hidden;
  const int *emk, *sidx, *smask, *tsi, *tsm, *tvi, *tvm;
  const float* slot_seed;
  const float *dyn1w, *dyn1b, *dyn2w, *dyn2b, *dynn_g, *dynn_b;
  const float *evp_w, *evp_b, *fbp_w, *fbp_b;
  const float *sk_w, *sk_b, *ek_w, *ek_b;
  const float *gate_w, *gate_b;
  const float *wd_w, *wd_b;
  const float *mq_w, *mq_b, *mk_w, *mk_b, *mv_w, *mv_b;
  const float *mn_g, *mn_b;
  const float *wp_w, *wp_b, *rp_w, *rp_b, *sym, *val;
  float *map_mem, *step_mem, *slots;
  float *scratch;
  size_t scr_stride;
};

__global__ __launch_bounds__(256) void chunk_kernel(ChunkP P) {
  const int b = blockIdx.x, t = threadIdx.x;
  float* mm = P.map_mem  + (size_t)b * NSYM_ * D_DIM;
  float* sme = P.step_mem + (size_t)b * NVAL_ * D_DIM;
  float* sl  = P.slots    + (size_t)b * NSLOT_ * D_DIM;
  float* scr = P.scratch  + (size_t)b * P.scr_stride;
  float* pred = scr;                               // 32*512
  float* t1   = pred + NSLOT_ * D_DIM;             // 32*512 (tanh(dyn1) then slot_key)
  float* cand = t1   + NSLOT_ * D_DIM;             // 32*512
  float* vb   = cand + NSLOT_ * D_DIM;             // 32*512
  float* qb   = vb   + NSLOT_ * D_DIM;             // 32*64
  float* kb   = qb   + NSLOT_ * 64;                // 32*64

  __shared__ float s_ev[512], s_w[512], s_h[512], s_fb[512], s_comb[512],
                   s_ek[512], s_row[512], s_attn[256], s_red[256];
  __shared__ float s_sc[32], s_ga[32], s_am[1024];
  __shared__ int s_flag[8];

  // init state (outputs live in d_out; must be deterministic across replays)
  for (int i = t; i < NSYM_ * D_DIM; i += 256) { mm[i] = 0.f; sme[i] = 0.f; }
  for (int i = t; i < NSLOT_ * D_DIM; i += 256) sl[i] = P.slot_seed[i];
  __syncthreads();

  for (int ci = 0; ci < NC_; ++ci) {
    const int s0 = ci * CHUNK_;
    // evidence = mean over chunk
    for (int d = t; d < D_DIM; d += 256) {
      float acc = 0.f;
      const float* hp = P.hidden + ((size_t)b * S_LEN + s0) * D_DIM + d;
      for (int s = 0; s < CHUNK_; ++s) acc += hp[(size_t)s * D_DIM];
      s_ev[d] = acc * (1.0f / CHUNK_);
    }
    if (t == 0) {
      int mk = 0, sm_ = 0, tsm_ = 0, tvm_ = 0;
      const int base = b * S_LEN + s0;
      for (int s = 0; s < CHUNK_; ++s) {
        mk  = (P.emk[base + s]   > mk)  ? P.emk[base + s]   : mk;
        sm_ = (P.smask[base + s] > sm_) ? P.smask[base + s] : sm_;
        tsm_ = (P.tsm[base + s] > tsm_) ? P.tsm[base + s] : tsm_;
        tvm_ = (P.tvm[base + s] > tvm_) ? P.tvm[base + s] : tvm_;
      }
      s_flag[0] = mk; s_flag[1] = P.sidx[base]; s_flag[2] = sm_;
      s_flag[3] = P.tsi[base]; s_flag[4] = tsm_;
      s_flag[5] = P.tvi[base]; s_flag[6] = tvm_;
    }
    __syncthreads();
    int marker = s_flag[0], src = s_flag[1];
    bool srcm = s_flag[2] != 0;
    int tgts = s_flag[3]; bool tgtsm = s_flag[4] != 0;
    int tgtv = s_flag[5]; bool tgtvm = s_flag[6] != 0;

    // graph update
    for (int d = t; d < D_DIM; d += 256) {
      float a = P.wp_b[d];
      for (int k = 0; k < D_DIM; ++k) a += s_ev[k] * P.wp_w[(size_t)k * D_DIM + d];
      s_w[d] = tanhf(a);
    }
    __syncthreads();
    bool g_src = srcm && (marker > 0);
    bool g_val = tgtsm && tgtvm;
    for (int d = t; d < D_DIM; d += 256) {
      if (g_src) mm[(size_t)src * D_DIM + d] += s_w[d];
      float msg = tanhf(s_w[d] + P.sym[(size_t)tgts * D_DIM + d]);
      if (g_val) sme[(size_t)tgtv * D_DIM + d] += msg;
    }
    __syncthreads();

    // graph call -> feedback
    graph_call_dev(mm, sme, src, srcm, P.rp_w, P.rp_b, s_h, s_attn, s_red, s_fb);

    // predicted = LN(slots + dyn2(tanh(dyn1(slots))))
    for (int i = t; i < NSLOT_ * D_DIM; i += 256) {
      int s_ = i >> 9, d = i & 511;
      float a = P.dyn1b[d];
      const float* row = sl + (size_t)s_ * D_DIM;
      for (int k = 0; k < D_DIM; ++k) a += row[k] * P.dyn1w[(size_t)k * D_DIM + d];
      t1[i] = tanhf(a);
    }
    __syncthreads();
    for (int i = t; i < NSLOT_ * D_DIM; i += 256) {
      int s_ = i >> 9, d = i & 511;
      float a = P.dyn2b[d];
      const float* row = t1 + (size_t)s_ * D_DIM;
      for (int k = 0; k < D_DIM; ++k) a += row[k] * P.dyn2w[(size_t)k * D_DIM + d];
      pred[i] = sl[i] + a;
    }
    __syncthreads();
    for (int r = 0; r < NSLOT_; ++r) {
      float* row = pred + (size_t)r * D_DIM;
      float v0 = row[t], v1 = row[t + 256];
      float m = blk_sum(v0 + v1, s_red) * (1.0f / D_DIM);
      float d0 = v0 - m, d1 = v1 - m;
      float var = blk_sum(d0 * d0 + d1 * d1, s_red) * (1.0f / D_DIM);
      float inv = rsqrtf(var + 1e-6f);
      row[t]       = d0 * inv * P.dynn_g[t] + P.dynn_b[t];
      row[t + 256] = d1 * inv * P.dynn_g[t + 256] + P.dynn_b[t + 256];
      __syncthreads();
    }

    // comb = ev_proj(ev) + fb_proj(fb); ek = ev_key(ev)
    for (int d = t; d < D_DIM; d += 256) {
      float a = P.evp_b[d], c2 = P.fbp_b[d], e = P.ek_b[d];
      for (int k = 0; k < D_DIM; ++k) {
        a  += s_ev[k] * P.evp_w[(size_t)k * D_DIM + d];
        c2 += s_fb[k] * P.fbp_w[(size_t)k * D_DIM + d];
        e  += s_ev[k] * P.ek_w[(size_t)k * D_DIM + d];
      }
      s_comb[d] = a + c2;
      s_ek[d] = e;
    }
    __syncthreads();

    // slot scores -> slot_attn
    for (int i = t; i < NSLOT_ * D_DIM; i += 256) {
      int s_ = i >> 9, d = i & 511;
      float a = P.sk_b[d];
      const float* row = pred + (size_t)s_ * D_DIM;
      for (int k = 0; k < D_DIM; ++k) a += row[k] * P.sk_w[(size_t)k * D_DIM + d];
      t1[i] = a;
    }
    __syncthreads();
    if (t < NSLOT_) {
      float a = 0.f;
      for (int d = 0; d < D_DIM; ++d) a += t1[(size_t)t * D_DIM + d] * s_ek[d];
      s_sc[t] = a * SCALE_D;
    }
    __syncthreads();
    if (t == 0) {
      float mx = -1e30f;
      for (int i = 0; i < NSLOT_; ++i) mx = fmaxf(mx, s_sc[i]);
      float sum = 0.f;
      for (int i = 0; i < NSLOT_; ++i) { s_sc[i] = __expf(s_sc[i] - mx); sum += s_sc[i]; }
      for (int i = 0; i < NSLOT_; ++i) s_sc[i] /= sum;
    }
    __syncthreads();

    // gates
    if (t < NSLOT_) {
      float a = P.gate_b[0];
      const float* row = pred + (size_t)t * D_DIM;
      for (int k = 0; k < D_DIM; ++k) a += row[k] * P.gate_w[k];
      for (int k = 0; k < D_DIM; ++k) a += s_comb[k] * P.gate_w[D_DIM + k];
      s_ga[t] = (1.0f / (1.0f + __expf(-a))) * s_sc[t];
    }
    __syncthreads();

    // cand = pred + gates * tanh(write_dense([pred, comb]))
    for (int i = t; i < NSLOT_ * D_DIM; i += 256) {
      int s_ = i >> 9, d = i & 511;
      float a = P.wd_b[d];
      const float* row = pred + (size_t)s_ * D_DIM;
      for (int k = 0; k < D_DIM; ++k) a += row[k] * P.wd_w[(size_t)k * D_DIM + d];
      for (int k = 0; k < D_DIM; ++k) a += s_comb[k] * P.wd_w[(size_t)(D_DIM + k) * D_DIM + d];
      cand[i] = pred[i] + s_ga[s_] * tanhf(a);
    }
    __syncthreads();

    // 3 explicit gradient steps
    for (int rs = 0; rs < 3; ++rs) {
      for (int d = t; d < D_DIM; d += 256) {
        float a = 0.f;
        for (int s_ = 0; s_ < NSLOT_; ++s_) a += cand[(size_t)s_ * D_DIM + d];
        s_row[d] = a * (1.0f / NSLOT_) - s_ev[d];
      }
      __syncthreads();
      const float c1 = 2.0f / ((float)BATCH_ * D_DIM * NSLOT_);
      for (int i = t; i < NSLOT_ * D_DIM; i += 256) {
        int d = i & 511;
        float gv = c1 * (s_row[d] + 0.35f * (cand[i] - pred[i]));
        cand[i] -= 0.05f * gv;
      }
      __syncthreads();
    }

    // low-rank mix attention
    for (int i = t; i < NSLOT_ * 64; i += 256) {
      int s_ = i >> 6, d = i & 63;
      const float* row = cand + (size_t)s_ * D_DIM;
      float aq = P.mq_b[d], ak = P.mk_b[d];
      for (int k = 0; k < D_DIM; ++k) {
        float r = row[k];
        aq += r * P.mq_w[(size_t)k * 64 + d];
        ak += r * P.mk_w[(size_t)k * 64 + d];
      }
      qb[i] = aq; kb[i] = ak;
    }
    for (int i = t; i < NSLOT_ * D_DIM; i += 256) {
      int s_ = i >> 9, d = i & 511;
      const float* row = cand + (size_t)s_ * D_DIM;
      float a = P.mv_b[d];
      for (int k = 0; k < D_DIM; ++k) a += row[k] * P.mv_w[(size_t)k * D_DIM + d];
      vb[i] = a;
    }
    __syncthreads();
    for (int i = t; i < NSLOT_ * NSLOT_; i += 256) {
      int s_ = i >> 5, tt = i & 31;
      float a = 0.f;
      for (int r = 0; r < 64; ++r) a += qb[(size_t)s_ * 64 + r] * kb[(size_t)tt * 64 + r];
      s_am[i] = a * 0.125f;
    }
    __syncthreads();
    if (t < NSLOT_) {
      float mx = -1e30f;
      for (int j = 0; j < NSLOT_; ++j) mx = fmaxf(mx, s_am[t * 32 + j]);
      float sum = 0.f;
      for (int j = 0; j < NSLOT_; ++j) { float e = __expf(s_am[t * 32 + j] - mx); s_am[t * 32 + j] = e; sum += e; }
      float inv = 1.0f / sum;
      for (int j = 0; j < NSLOT_; ++j) s_am[t * 32 + j] *= inv;
    }
    __syncthreads();
    for (int r = 0; r < NSLOT_; ++r) {
      for (int d = t; d < D_DIM; d += 256) {
        float a = 0.f;
        for (int tt = 0; tt < NSLOT_; ++tt) a += s_am[r * 32 + tt] * vb[(size_t)tt * D_DIM + d];
        s_row[d] = cand[(size_t)r * D_DIM + d] + a;
      }
      __syncthreads();
      float v0 = s_row[t], v1 = s_row[t + 256];
      float m = blk_sum(v0 + v1, s_red) * (1.0f / D_DIM);
      float d0 = v0 - m, d1 = v1 - m;
      float var = blk_sum(d0 * d0 + d1 * d1, s_red) * (1.0f / D_DIM);
      float inv = rsqrtf(var + 1e-6f);
      sl[(size_t)r * D_DIM + t]       = d0 * inv * P.mn_g[t] + P.mn_b[t];
      sl[(size_t)r * D_DIM + t + 256] = d1 * inv * P.mn_g[t + 256] + P.mn_b[t + 256];
      __syncthreads();
    }
  }
}

// ---------------------------------------------------------------------------
// Final readout: graph_call(query) + slot attention readout + answer head
// ---------------------------------------------------------------------------
struct FinalP {
  const float* hidden;
  const int* lengths;
  const int* qidx;
  const unsigned char* qmask;
  const float *rp_w, *rp_b, *val;
  const float *map_mem, *step_mem, *slots;
  const float *ans1w, *ans1b, *ans2w, *ans2b;
  float *out_ans, *out_glog;
};

__global__ __launch_bounds__(256) void final_kernel(FinalP P) {
  int b = blockIdx.x, t = threadIdx.x;
  __shared__ float s_ac[512], s_h[512], s_attn[256], s_red[256], s_fb[512],
                   s_cat[1536], s_a1[512], s_sc[32];
  const float* mm = P.map_mem + (size_t)b * NSYM_ * D_DIM;
  const float* sme = P.step_mem + (size_t)b * NVAL_ * D_DIM;
  const float* sl = P.slots + (size_t)b * NSLOT_ * D_DIM;

  int len = P.lengths[b];
  int row = len - 1; row = row < 0 ? 0 : (row > S_LEN - 1 ? S_LEN - 1 : row);
  for (int d = t; d < D_DIM; d += 256)
    s_ac[d] = P.hidden[((size_t)b * S_LEN + row) * D_DIM + d];
  __syncthreads();

  graph_call_dev(mm, sme, P.qidx[b], P.qmask[b] != 0, P.rp_w, P.rp_b,
                 s_h, s_attn, s_red, s_fb);

  float gl;
  {
    float a = 0.f;
    const float* vrow = P.val + (size_t)t * D_DIM;
    for (int k = 0; k < D_DIM; ++k) a += s_fb[k] * vrow[k];
    gl = a;
    P.out_glog[b * NVAL_ + t] = a;
  }
  if (t < NSLOT_) {
    float a = 0.f;
    for (int d = 0; d < D_DIM; ++d) a += sl[(size_t)t * D_DIM + d] * s_ac[d];
    s_sc[t] = a * SCALE_D;
  }
  __syncthreads();
  if (t == 0) {
    float mx = -1e30f;
    for (int i = 0; i < NSLOT_; ++i) mx = fmaxf(mx, s_sc[i]);
    float sum = 0.f;
    for (int i = 0; i < NSLOT_; ++i) { s_sc[i] = __expf(s_sc[i] - mx); sum += s_sc[i]; }
    for (int i = 0; i < NSLOT_; ++i) s_sc[i] /= sum;
  }
  __syncthreads();
  for (int d = t; d < D_DIM; d += 256) {
    float a = 0.f;
    for (int s_ = 0; s_ < NSLOT_; ++s_) a += s_sc[s_] * sl[(size_t)s_ * D_DIM + d];
    s_cat[d] = s_ac[d];
    s_cat[D_DIM + d] = a;
    s_cat[2 * D_DIM + d] = s_fb[d];
  }
  __syncthreads();
  for (int d = t; d < D_DIM; d += 256) {
    float a = P.ans1b[d];
    for (int k = 0; k < 3 * D_DIM; ++k) a += s_cat[k] * P.ans1w[(size_t)k * D_DIM + d];
    s_a1[d] = gelu_f(a);
  }
  __syncthreads();
  {
    float a = P.ans2b[t];
    for (int k = 0; k < D_DIM; ++k) a += s_a1[k] * P.ans2w[(size_t)k * NVAL_ + t];
    P.out_ans[b * NVAL_ + t] = a + 2.0f * gl;
  }
}

// ---------------------------------------------------------------------------
// Host launcher
// ---------------------------------------------------------------------------
// params pytree flattens after the 11 input arrays, dict keys sorted:
enum {
  P_ANS1_B = 11, P_ANS1_W, P_ANS2_B, P_ANS2_W,
  P_CONV1_B, P_CONV1_W, P_CONV2_B, P_CONV2_W,
  P_DYN1_B, P_DYN1_W, P_DYN2_B, P_DYN2_W,
  P_DYNN_B, P_DYNN_G, P_ENCN_B, P_ENCN_G,
  P_EVK_B, P_EVK_W, P_EVP_B, P_EVP_W, P_FBP_B, P_FBP_W,
  P_GATE_B, P_GATE_W,
  P_RP_B, P_RP_W, P_SYME, P_VALE, P_WP_B, P_WP_W,
  P_LM_B, P_LM_W,
  P_MIXK_B, P_MIXK_W, P_MIXN_B, P_MIXN_G, P_MIXQ_B, P_MIXQ_W, P_MIXV_B, P_MIXV_W,
  P_POSE, P_SLOTK_B, P_SLOTK_W, P_SLOTSEED, P_TOKE, P_WD_B, P_WD_W
};

extern "C" void kernel_launch(void* const* d_in, const int* in_sizes, int n_in,
                              void* d_out, int out_size, void* d_ws, size_t ws_size,
                              hipStream_t stream) {
  (void)in_sizes; (void)n_in; (void)out_size; (void)ws_size;
  const int* tok      = (const int*)d_in[0];
  const int* lengths  = (const int*)d_in[1];
  const int* qidx     = (const int*)d_in[2];
  const unsigned char* qmask = (const unsigned char*)d_in[3];
  const int* emk      = (const int*)d_in[4];
  const int* sidx     = (const int*)d_in[5];
  const int* smask    = (const int*)d_in[6];
  const int* tsi      = (const int*)d_in[7];
  const int* tsm      = (const int*)d_in[8];
  const int* tvi      = (const int*)d_in[9];
  const int* tvm      = (const int*)d_in[10];
  const float* F = nullptr; (void)F;
  #define PF(i) ((const float*)d_in[(i)])

  // workspace carve-up
  char* ws = (char*)d_ws;
  size_t off = 0;
  auto take = [&](size_t bytes) -> char* {
    char* r = ws + off;
    off = (off + bytes + 255) & ~(size_t)255;
    return r;
  };
  const size_t MROWS = (size_t)BATCH_ * S_LEN;
  bf16_t* xb      = (bf16_t*)take(MROWS * D_DIM * 2);
  bf16_t* h1b     = (bf16_t*)take(MROWS * DEXP_ * 2);
  float*  h2      = (float*) take(MROWS * D_DIM * 4);
  float*  hidden  = (float*) take(MROWS * D_DIM * 4);
  bf16_t* hiddenb = (bf16_t*)take(MROWS * D_DIM * 2);
  bf16_t* wt1     = (bf16_t*)take((size_t)5 * DEXP_ * D_DIM * 2);
  bf16_t* wt2     = (bf16_t*)take((size_t)5 * D_DIM * DEXP_ * 2);
  bf16_t* wtlm    = (bf16_t*)take((size_t)V_DIM * D_DIM * 2);
  const size_t SCR_STRIDE = (size_t)NSLOT_ * D_DIM * 4 + (size_t)NSLOT_ * 64 * 2;
  float*  scratch = (float*) take((size_t)BATCH_ * SCR_STRIDE * 4);

  // outputs (concatenated flat in reference return order)
  float* out_lm    = (float*)d_out;
  float* out_ans   = out_lm   + MROWS * (size_t)V_DIM;
  float* out_glog  = out_ans  + (size_t)BATCH_ * NVAL_;
  float* out_map   = out_glog + (size_t)BATCH_ * NVAL_;
  float* out_step  = out_map  + (size_t)BATCH_ * NSYM_ * D_DIM;
  float* out_slots = out_step + (size_t)BATCH_ * NVAL_ * D_DIM;

  dim3 blk(256);

  // 1) weight transposes -> bf16 WT[n][k]
  for (int t = 0; t < 5; ++t) {
    size_t n1 = (size_t)D_DIM * DEXP_;
    transpose_bf16_kernel<<<dim3((n1 + 255) / 256), blk, 0, stream>>>(
        PF(P_CONV1_W) + (size_t)t * D_DIM * DEXP_, wt1 + (size_t)t * DEXP_ * D_DIM, D_DIM, DEXP_);
    transpose_bf16_kernel<<<dim3((n1 + 255) / 256), blk, 0, stream>>>(
        PF(P_CONV2_W) + (size_t)t * DEXP_ * D_DIM, wt2 + (size_t)t * D_DIM * DEXP_, DEXP_, D_DIM);
  }
  {
    size_t n1 = (size_t)D_DIM * V_DIM;
    transpose_bf16_kernel<<<dim3((n1 + 255) / 256), blk, 0, stream>>>(
        PF(P_LM_W), wtlm, D_DIM, V_DIM);
  }

  // 2) embedding
  embed_kernel<<<dim3((unsigned)MROWS), blk, 0, stream>>>(tok, PF(P_TOKE), PF(P_POSE), xb);

  // 3) conv1 (+GELU, bf16 out)
  conv_gemm_kernel<<<dim3(MROWS / 128, DEXP_ / 64), blk, 0, stream>>>(
      xb, wt1, PF(P_CONV1_B), nullptr, h1b, D_DIM, DEXP_, 1);

  // 4) conv2 (f32 out, pre-LN)
  conv_gemm_kernel<<<dim3(MROWS / 128, D_DIM / 64), blk, 0, stream>>>(
      h1b, wt2, PF(P_CONV2_B), h2, nullptr, DEXP_, D_DIM, 0);

  // 5) encoder layernorm
  ln_kernel<<<dim3((unsigned)MROWS), blk, 0, stream>>>(
      h2, PF(P_ENCN_G), PF(P_ENCN_B), hidden, hiddenb);

  // 6) fused chunk recurrence (per batch)
  ChunkP cp;
  cp.hidden = hidden;
  cp.emk = emk; cp.sidx = sidx; cp.smask = smask;
  cp.tsi = tsi; cp.tsm = tsm; cp.tvi = tvi; cp.tvm = tvm;
  cp.slot_seed = PF(P_SLOTSEED);
  cp.dyn1w = PF(P_DYN1_W); cp.dyn1b = PF(P_DYN1_B);
  cp.dyn2w = PF(P_DYN2_W); cp.dyn2b = PF(P_DYN2_B);
  cp.dynn_g = PF(P_DYNN_G); cp.dynn_b = PF(P_DYNN_B);
  cp.evp_w = PF(P_EVP_W); cp.evp_b = PF(P_EVP_B);
  cp.fbp_w = PF(P_FBP_W); cp.fbp_b = PF(P_FBP_B);
  cp.sk_w = PF(P_SLOTK_W); cp.sk_b = PF(P_SLOTK_B);
  cp.ek_w = PF(P_EVK_W); cp.ek_b = PF(P_EVK_B);
  cp.gate_w = PF(P_GATE_W); cp.gate_b = PF(P_GATE_B);
  cp.wd_w = PF(P_WD_W); cp.wd_b = PF(P_WD_B);
  cp.mq_w = PF(P_MIXQ_W); cp.mq_b = PF(P_MIXQ_B);
  cp.mk_w = PF(P_MIXK_W); cp.mk_b = PF(P_MIXK_B);
  cp.mv_w = PF(P_MIXV_W); cp.mv_b = PF(P_MIXV_B);
  cp.mn_g = PF(P_MIXN_G); cp.mn_b = PF(P_MIXN_B);
  cp.wp_w = PF(P_WP_W); cp.wp_b = PF(P_WP_B);
  cp.rp_w = PF(P_RP_W); cp.rp_b = PF(P_RP_B);
  cp.sym = PF(P_SYME); cp.val = PF(P_VALE);
  cp.map_mem = out_map; cp.step_mem = out_step; cp.slots = out_slots;
  cp.scratch = scratch; cp.scr_stride = SCR_STRIDE;
  chunk_kernel<<<dim3(BATCH_), blk, 0, stream>>>(cp);

  // 7) lm_head GEMM (WMMA bf16, store-bound)
  gemm_lm_kernel<<<dim3(MROWS / 128, V_DIM / 64), blk, 0, stream>>>(
      hiddenb, wtlm, PF(P_LM_B), out_lm, (int)MROWS, V_DIM, D_DIM);

  // 8) final readout / answer head
  FinalP fp;
  fp.hidden = hidden; fp.lengths = lengths; fp.qidx = qidx; fp.qmask = qmask;
  fp.rp_w = PF(P_RP_W); fp.rp_b = PF(P_RP_B); fp.val = PF(P_VALE);
  fp.map_mem = out_map; fp.step_mem = out_step; fp.slots = out_slots;
  fp.ans1w = PF(P_ANS1_W); fp.ans1b = PF(P_ANS1_B);
  fp.ans2w = PF(P_ANS2_W); fp.ans2b = PF(P_ANS2_B);
  fp.out_ans = out_ans; fp.out_glog = out_glog;
  final_kernel<<<dim3(BATCH_), blk, 0, stream>>>(fp);
  #undef PF
}
